// Self_ST_16415365005892
// MI455X (gfx1250) — matrix operand order repaired and checked by reading, not compiled, not run on hardware
//
#include <hip/hip_runtime.h>

#define B_  8
#define C_  256
#define N_  9216
#define NS_ 1024
#define HW_ 96

typedef __bf16 bf16x16 __attribute__((ext_vector_type(16)));
typedef float  f32x8   __attribute__((ext_vector_type(8)));

union BFrag {
    bf16x16 f;
    uint4   q[2];
};

__device__ __forceinline__ unsigned short f2bf(float x) {
    return __builtin_bit_cast(unsigned short, (__bf16)x);   // hw v_cvt bf16
}

__device__ __forceinline__ f32x8 wmma_bf16(bf16x16 a, bf16x16 b, f32x8 c) {
    return __builtin_amdgcn_wmma_f32_16x16x32_bf16(false, a, false, b, (short)0, c, false, false);
}

// CDNA5 async global->LDS copy (ASYNCcnt-tracked, no VGPR round trip).
__device__ __forceinline__ void async_b128(void* lds, const void* g) {
    unsigned off = (unsigned)(size_t)lds;        // low 32 bits of generic ptr = LDS byte offset
    asm volatile("global_load_async_to_lds_b128 %0, %1, off"
                 :: "v"(off), "v"(g) : "memory");
}
__device__ __forceinline__ void wait_async0() {
    asm volatile("s_wait_asynccnt 0x0" ::: "memory");
}

// ---------------------------------------------------------------------------
// Kernel 0: fold weights.  W_ks = Ws*Wk, W_vs = Ws*Wv, b_ks = Ws*bk + bs, etc.
// ---------------------------------------------------------------------------
__global__ void k_fold(const float* __restrict__ Wq, const float* __restrict__ Wk,
                       const float* __restrict__ Wv, const float* __restrict__ Ws,
                       const float* __restrict__ bk, const float* __restrict__ bv,
                       const float* __restrict__ bs,
                       unsigned short* __restrict__ Wqb, unsigned short* __restrict__ Wks,
                       unsigned short* __restrict__ Wvs,
                       float* __restrict__ bks, float* __restrict__ bvs) {
    int o = blockIdx.x, c = threadIdx.x;
    float sk = 0.f, sv = 0.f;
    for (int t = 0; t < C_; ++t) {
        float w = Ws[o * C_ + t];
        sk += w * Wk[t * C_ + c];
        sv += w * Wv[t * C_ + c];
    }
    Wks[o * C_ + c] = f2bf(sk);
    Wvs[o * C_ + c] = f2bf(sv);
    Wqb[o * C_ + c] = f2bf(Wq[o * C_ + c]);
    if (c == 0) {
        float sbk = 0.f, sbv = 0.f;
        for (int t = 0; t < C_; ++t) { float w = Ws[o * C_ + t]; sbk += w * bk[t]; sbv += w * bv[t]; }
        bks[o] = sbk + bs[o];
        bvs[o] = sbv + bs[o];
    }
}

// ---------------------------------------------------------------------------
// Kernel 1: Q = Wq * X + bq -> Qbuf[b][n][cq] bf16 (A-fragment layout).
// Wq chunks staged into LDS with async loads; X transposed via VGPR path
// (needs f32->bf16 conversion on the way in).
// ---------------------------------------------------------------------------
__global__ void __launch_bounds__(256) k_q(const float* __restrict__ x,
                                           const unsigned short* __restrict__ Wqb,
                                           const float* __restrict__ bq,
                                           unsigned short* __restrict__ Qbuf) {
    __shared__ unsigned short Xt[128 * 264];   // [n_local][c]
    __shared__ unsigned short Wst[256 * 40];   // [cq][c_chunk 32]

    const int tid  = threadIdx.x;
    const int wv   = tid >> 5, ln = tid & 31;
    const int half = ln >> 4,  l16 = ln & 15;
    const int b    = blockIdx.y;
    const int n0   = blockIdx.x * 128;

    for (int i = tid; i < 128 * 256; i += 256) {
        int c = i >> 7, j = i & 127;
        Xt[j * 264 + c] = f2bf(x[((size_t)b * C_ + c) * N_ + n0 + j]);
    }

    f32x8 acc[16];
#pragma unroll
    for (int t = 0; t < 16; ++t) acc[t] = (f32x8)0.0f;

    for (int k = 0; k < 8; ++k) {
        __syncthreads();                       // previous chunk consumed
        for (int i = tid; i < 1024; i += 256) {
            int rr = i >> 2, c4 = i & 3;
            async_b128(&Wst[rr * 40 + c4 * 8], Wqb + rr * C_ + k * 32 + c4 * 8);
        }
        wait_async0();
        __syncthreads();

        BFrag a;
        {
            const unsigned short* p = &Xt[(wv * 16 + l16) * 264 + k * 32 + half * 8];
            a.q[0] = *(const uint4*)p;
            a.q[1] = *(const uint4*)(p + 16);
        }
#pragma unroll
        for (int ct = 0; ct < 16; ++ct) {
            BFrag bw;
            const unsigned short* p = &Wst[(ct * 16 + l16) * 40 + half * 8];
            bw.q[0] = *(const uint4*)p;
            bw.q[1] = *(const uint4*)(p + 16);
            acc[ct] = wmma_bf16(a.f, bw.f, acc[ct]);
        }
    }

#pragma unroll
    for (int ct = 0; ct < 16; ++ct) {
        float bqv = bq[ct * 16 + l16];
#pragma unroll
        for (int r = 0; r < 8; ++r) {
            int n = n0 + wv * 16 + r + 8 * half;
            Qbuf[((size_t)b * N_ + n) * C_ + ct * 16 + l16] = f2bf(acc[ct][r] + bqv);
        }
    }
}

// ---------------------------------------------------------------------------
// Kernel 2: K,V on the stride-3 subsample with the folded weights.
//   Kbuf[b][m][ck]  (ck contiguous -> B-operand layout for S GEMM)
//   Vbuf[b][cv][m]  (m  contiguous -> B-operand layout for O GEMM)
// ---------------------------------------------------------------------------
__global__ void __launch_bounds__(256) k_kv(const float* __restrict__ x,
                                            const unsigned short* __restrict__ Wks,
                                            const unsigned short* __restrict__ Wvs,
                                            const float* __restrict__ bks,
                                            const float* __restrict__ bvs,
                                            unsigned short* __restrict__ Kbuf,
                                            unsigned short* __restrict__ Vbuf) {
    __shared__ unsigned short Xst[128 * 264];

    const int tid  = threadIdx.x;
    const int wv   = tid >> 5, ln = tid & 31;
    const int half = ln >> 4,  l16 = ln & 15;
    const int b    = blockIdx.y;
    const int m0   = blockIdx.x * 128;

    for (int i = tid; i < 128 * 256; i += 256) {
        int c = i >> 7, j = i & 127;
        int m = m0 + j;
        int ns = (m >> 5) * 3 * HW_ + (m & 31) * 3;
        Xst[j * 264 + c] = f2bf(x[((size_t)b * C_ + c) * N_ + ns]);
    }
    __syncthreads();

    // K phase: D(M=m, N=ck); A = Xst, B = Wks (L2-resident)
    {
        f32x8 acc[16];
#pragma unroll
        for (int t = 0; t < 16; ++t) acc[t] = (f32x8)0.0f;
        for (int k = 0; k < 8; ++k) {
            BFrag a;
            const unsigned short* p = &Xst[(wv * 16 + l16) * 264 + k * 32 + half * 8];
            a.q[0] = *(const uint4*)p;
            a.q[1] = *(const uint4*)(p + 16);
#pragma unroll
            for (int ct = 0; ct < 16; ++ct) {
                BFrag bw;
                const unsigned short* w = Wks + (ct * 16 + l16) * C_ + k * 32 + half * 8;
                bw.q[0] = *(const uint4*)w;
                bw.q[1] = *(const uint4*)(w + 16);
                acc[ct] = wmma_bf16(a.f, bw.f, acc[ct]);
            }
        }
#pragma unroll
        for (int ct = 0; ct < 16; ++ct) {
            float bv_ = bks[ct * 16 + l16];
#pragma unroll
            for (int r = 0; r < 8; ++r) {
                int m = m0 + wv * 16 + r + 8 * half;
                Kbuf[((size_t)b * NS_ + m) * C_ + ct * 16 + l16] = f2bf(acc[ct][r] + bv_);
            }
        }
    }

    // V phase: D(M=cv, N=m); A = Wvs, B = Xst
    {
        f32x8 accv[16];
#pragma unroll
        for (int t = 0; t < 16; ++t) accv[t] = (f32x8)0.0f;
        for (int k = 0; k < 8; ++k) {
            BFrag a0, a1;
            {
                const unsigned short* w0 = Wvs + (wv * 16 + l16) * C_ + k * 32 + half * 8;
                a0.q[0] = *(const uint4*)w0;  a0.q[1] = *(const uint4*)(w0 + 16);
                const unsigned short* w1 = Wvs + ((wv + 8) * 16 + l16) * C_ + k * 32 + half * 8;
                a1.q[0] = *(const uint4*)w1;  a1.q[1] = *(const uint4*)(w1 + 16);
            }
#pragma unroll
            for (int mt = 0; mt < 8; ++mt) {
                BFrag bx;
                const unsigned short* p = &Xst[(mt * 16 + l16) * 264 + k * 32 + half * 8];
                bx.q[0] = *(const uint4*)p;
                bx.q[1] = *(const uint4*)(p + 16);
                accv[mt]     = wmma_bf16(a0.f, bx.f, accv[mt]);
                accv[8 + mt] = wmma_bf16(a1.f, bx.f, accv[8 + mt]);
            }
        }
        float bvv[16];
#pragma unroll
        for (int t = 0; t < 16; ++t)
            bvv[t] = bvs[(wv + 8 * (t >> 3)) * 16 + (t & 7) + 8 * half];
#pragma unroll
        for (int c2 = 0; c2 < 2; ++c2)
#pragma unroll
            for (int mt = 0; mt < 8; ++mt)
#pragma unroll
                for (int r = 0; r < 8; ++r) {
                    int cv = (wv + 8 * c2) * 16 + r + 8 * half;
                    Vbuf[((size_t)b * C_ + cv) * NS_ + m0 + mt * 16 + l16] =
                        f2bf(accv[c2 * 8 + mt][r] + bvv[c2 * 8 + r]);
                }
    }
}

// ---------------------------------------------------------------------------
// Kernel 3: fused flash attention, async double-buffered K/V staging.
// 128 query rows per workgroup (8 waves x 16 rows), m processed in chunks of
// 32 with ping-pong LDS buffers filled by global_load_async_to_lds_b128 while
// WMMAs consume the other buffer.  Pass 1: row max.  Pass 2: recompute S,
// P = exp(S-max), row sums, O += P*V^T.  Epilogue transposed through LDS.
// ---------------------------------------------------------------------------
__global__ void __launch_bounds__(256) k_attn(const float* __restrict__ x,
                                              const unsigned short* __restrict__ Qbuf,
                                              const unsigned short* __restrict__ Kbuf,
                                              const unsigned short* __restrict__ Vbuf,
                                              const float* __restrict__ gamma,
                                              float* __restrict__ out) {
    __shared__ unsigned short Kst[2][32 * 264];   // [buf][m_local 32][c 256]
    __shared__ unsigned short Vst[2][256 * 40];   // [buf][c 256][m_local 32]
    __shared__ unsigned short Pst[8 * 16 * 40];   // per-wave P tile [16 n][32 m]
    __shared__ float          Obl[16 * 132];      // [c_local 16][n_local 128]

    const int tid  = threadIdx.x;
    const int wv   = tid >> 5, ln = tid & 31;
    const int half = ln >> 4,  l16 = ln & 15;
    const int b    = blockIdx.y;
    const int n0   = blockIdx.x * 128;

    // persistent Q A-fragments (16 rows x 256 cq)
    BFrag aq[8];
    {
        const unsigned short* qr = Qbuf + ((size_t)b * N_ + n0 + wv * 16 + l16) * C_;
#pragma unroll
        for (int k = 0; k < 8; ++k) {
            const unsigned short* p = qr + k * 32 + half * 8;
            aq[k].q[0] = *(const uint4*)p;
            aq[k].q[1] = *(const uint4*)(p + 16);
        }
    }

    const unsigned short* Kb = Kbuf + (size_t)b * NS_ * C_;
    const unsigned short* Vb = Vbuf + (size_t)b * C_ * NS_;

    // chunk staging: 1024 b128 each for K and V, 4 per thread
#define STAGE_K(buf, cc)                                                      \
    for (int t_ = 0; t_ < 4; ++t_) {                                          \
        int i_ = tid + t_ * 256;                                              \
        int row_ = i_ >> 5, col_ = i_ & 31;                                   \
        async_b128(&Kst[buf][row_ * 264 + col_ * 8],                          \
                   Kb + ((size_t)(cc) * 32 + row_) * C_ + col_ * 8);          \
    }
#define STAGE_V(buf, cc)                                                      \
    for (int t_ = 0; t_ < 4; ++t_) {                                          \
        int i_ = tid + t_ * 256;                                              \
        int row_ = i_ >> 2, col_ = i_ & 3;                                    \
        async_b128(&Vst[buf][row_ * 40 + col_ * 8],                           \
                   Vb + (size_t)row_ * NS_ + (cc) * 32 + col_ * 8);           \
    }

    float vmax[8];
#pragma unroll
    for (int r = 0; r < 8; ++r) vmax[r] = -3.0e38f;

    // ----------------- pass 1: row max -----------------
    STAGE_K(0, 0)
    wait_async0();
    __syncthreads();
    for (int cc = 0; cc < 32; ++cc) {
        int cur = cc & 1;
        if (cc + 1 < 32) { STAGE_K(cur ^ 1, cc + 1) }
#pragma unroll
        for (int mt = 0; mt < 2; ++mt) {
            f32x8 s = (f32x8)0.0f;
#pragma unroll
            for (int k = 0; k < 8; ++k) {
                BFrag bk_;
                const unsigned short* p = &Kst[cur][(mt * 16 + l16) * 264 + k * 32 + half * 8];
                bk_.q[0] = *(const uint4*)p;
                bk_.q[1] = *(const uint4*)(p + 16);
                s = wmma_bf16(aq[k].f, bk_.f, s);
            }
#pragma unroll
            for (int r = 0; r < 8; ++r) vmax[r] = fmaxf(vmax[r], s[r]);
        }
        wait_async0();
        __syncthreads();
    }
#pragma unroll
    for (int off = 8; off >= 1; off >>= 1)
#pragma unroll
        for (int r = 0; r < 8; ++r)
            vmax[r] = fmaxf(vmax[r], __shfl_xor(vmax[r], off, 32));

    // ----------------- pass 2: exp + O accumulation -----------------
    float vsum[8];
    f32x8 Oa[16];
#pragma unroll
    for (int r = 0; r < 8; ++r) vsum[r] = 0.f;
#pragma unroll
    for (int t = 0; t < 16; ++t) Oa[t] = (f32x8)0.0f;

    STAGE_K(0, 0)
    STAGE_V(0, 0)
    wait_async0();
    __syncthreads();
    for (int cc = 0; cc < 32; ++cc) {
        int cur = cc & 1;
        if (cc + 1 < 32) { STAGE_K(cur ^ 1, cc + 1) STAGE_V(cur ^ 1, cc + 1) }

#pragma unroll
        for (int mt = 0; mt < 2; ++mt) {               // recompute S, stage P
            f32x8 s = (f32x8)0.0f;
#pragma unroll
            for (int k = 0; k < 8; ++k) {
                BFrag bk_;
                const unsigned short* p = &Kst[cur][(mt * 16 + l16) * 264 + k * 32 + half * 8];
                bk_.q[0] = *(const uint4*)p;
                bk_.q[1] = *(const uint4*)(p + 16);
                s = wmma_bf16(aq[k].f, bk_.f, s);
            }
#pragma unroll
            for (int r = 0; r < 8; ++r) {
                float pv = __expf(s[r] - vmax[r]);
                vsum[r] += pv;
                Pst[wv * 640 + (r + 8 * half) * 40 + mt * 16 + l16] = f2bf(pv);
            }
        }
        BFrag pa;                                      // P as A-frag (K = 32 m)
        {
            const unsigned short* p = &Pst[wv * 640 + l16 * 40 + half * 8];
            pa.q[0] = *(const uint4*)p;
            pa.q[1] = *(const uint4*)(p + 16);
        }
#pragma unroll
        for (int ct = 0; ct < 16; ++ct) {              // O += P * V^T
            BFrag bv_;
            const unsigned short* p = &Vst[cur][(ct * 16 + l16) * 40 + half * 8];
            bv_.q[0] = *(const uint4*)p;
            bv_.q[1] = *(const uint4*)(p + 16);
            Oa[ct] = wmma_bf16(pa.f, bv_.f, Oa[ct]);
        }
        wait_async0();
        __syncthreads();
    }

#pragma unroll
    for (int off = 8; off >= 1; off >>= 1)
#pragma unroll
        for (int r = 0; r < 8; ++r)
            vsum[r] += __shfl_xor(vsum[r], off, 32);
    float rinv[8];
#pragma unroll
    for (int r = 0; r < 8; ++r) rinv[r] = 1.0f / vsum[r];

    // ----------------- epilogue: gamma * O/sum + x -----------------
    const float gm = gamma[0];
    for (int ct = 0; ct < 16; ++ct) {
        __syncthreads();
#pragma unroll
        for (int r = 0; r < 8; ++r)
            Obl[l16 * 132 + wv * 16 + r + 8 * half] = Oa[ct][r] * rinv[r];
        __syncthreads();
        for (int i = tid; i < 512; i += 256) {         // 512 float4 = 16c x 128n
            int cl = i >> 5, n4 = i & 31;
            float4 o = *(const float4*)&Obl[cl * 132 + n4 * 4];
            size_t gi = ((size_t)b * C_ + ct * 16 + cl) * N_ + n0 + n4 * 4;
            float4 xv = *(const float4*)(x + gi);
            float4 res;
            res.x = gm * o.x + xv.x;  res.y = gm * o.y + xv.y;
            res.z = gm * o.z + xv.z;  res.w = gm * o.w + xv.w;
            *(float4*)(out + gi) = res;
        }
    }
#undef STAGE_K
#undef STAGE_V
}

// ---------------------------------------------------------------------------
extern "C" void kernel_launch(void* const* d_in, const int* in_sizes, int n_in,
                              void* d_out, int out_size, void* d_ws, size_t ws_size,
                              hipStream_t stream) {
    (void)in_sizes; (void)n_in; (void)out_size; (void)ws_size;
    const float* x  = (const float*)d_in[0];
    const float* Wq = (const float*)d_in[1];
    const float* bq = (const float*)d_in[2];
    const float* Wk = (const float*)d_in[3];
    const float* bk = (const float*)d_in[4];
    const float* Wv = (const float*)d_in[5];
    const float* bv = (const float*)d_in[6];
    const float* Ws = (const float*)d_in[7];
    const float* bs = (const float*)d_in[8];
    const float* gm = (const float*)d_in[9];
    float* out = (float*)d_out;

    char* ws = (char*)d_ws;                 // ~46.5 MB total
    unsigned short* Qbuf = (unsigned short*)(ws);
    unsigned short* Kbuf = (unsigned short*)(ws + 37748736);
    unsigned short* Vbuf = (unsigned short*)(ws + 37748736 + 4194304);
    char* wbase = ws + 37748736 + 2 * 4194304;
    unsigned short* Wqb  = (unsigned short*)(wbase);
    unsigned short* Wks  = (unsigned short*)(wbase + 131072);
    unsigned short* Wvs  = (unsigned short*)(wbase + 2 * 131072);
    float*          bks  = (float*)(wbase + 3 * 131072);
    float*          bvs  = (float*)(wbase + 3 * 131072 + 4096);

    k_fold<<<dim3(256), dim3(256), 0, stream>>>(Wq, Wk, Wv, Ws, bk, bv, bs,
                                                Wqb, Wks, Wvs, bks, bvs);
    k_q   <<<dim3(72, 8), dim3(256), 0, stream>>>(x, Wqb, bq, Qbuf);
    k_kv  <<<dim3(8, 8),  dim3(256), 0, stream>>>(x, Wks, Wvs, bks, bvs, Kbuf, Vbuf);
    k_attn<<<dim3(72, 8), dim3(256), 0, stream>>>(x, Qbuf, Kbuf, Vbuf, gm, out);
}
